// MambaBranch_87935160418832
// MI455X (gfx1250) — compile-verified, matrix operand order checked by reading, not run on hardware
//
#include <hip/hip_runtime.h>

typedef __bf16 bf16;
typedef __attribute__((ext_vector_type(16))) __bf16 v16bf;
typedef __attribute__((ext_vector_type(8)))  float  v8f;
typedef unsigned int u32x4 __attribute__((ext_vector_type(4)));
typedef int          i32x8 __attribute__((ext_vector_type(8)));
typedef int          i32x4 __attribute__((ext_vector_type(4)));

#define BM 64
#define BN 64
#define BK 32
#define LN_EPS 1e-5f

// Tensor Data Mover available? (device pass only; host pass -> 0 -> fallback)
#if defined(__AMDGCN__) && __has_builtin(__builtin_amdgcn_tensor_load_to_lds) && __has_builtin(__builtin_amdgcn_s_wait_tensorcnt)
#define USE_TDM 1
#else
#define USE_TDM 0
#endif

__device__ __forceinline__ float gelu_f(float v) {
    return 0.5f * v * (1.0f + erff(v * 0.70710678118654752f));
}

// ---------------------------------------------------------------------------
// WMMA GEMM:  C[z] = A(MxK, bf16, shared) * B[z](KxN, bf16)  (+bias)(+act)(+res)
// 256 threads = 8 waves in a 2(M)x4(N) grid over a 64x64 block tile; each wave
// owns a 32x16 C strip -> two v_wmma_f32_16x16x32_bf16 per K-slab sharing one
// B fragment. A tile staged into LDS by the Tensor Data Mover (2D descriptor,
// HW zero-fill on M/K edges) when available; B tile staged transposed so the
// per-lane fragment gathers are two ds_load_b128 each.
// ---------------------------------------------------------------------------
__global__ void __launch_bounds__(256) gemm_wmma_kernel(
    const bf16* __restrict__ A, const bf16* __restrict__ Bmat,
    float* __restrict__ C, bf16* __restrict__ Cb,
    const float* __restrict__ bias, const float* __restrict__ res,
    int M, int N, int K, long long sB, long long sC, int act)
{
    __shared__ bf16 As[BM * BK];     // row-major [64][32]
    __shared__ bf16 Bt[BN * BK];     // transposed [n][k] so K-pairs are contiguous

    const bf16* Bp   = Bmat + (long long)blockIdx.z * sB;
    long long   coff = (long long)blockIdx.z * sC;
    int bm = blockIdx.y * BM, bn = blockIdx.x * BN;
    int tid  = threadIdx.x;
    int wave = tid >> 5, lane = tid & 31;
    int wm = wave >> 2, wn = wave & 3;           // 2x4 wave grid
    int hi16 = lane >> 4, l15 = lane & 15;

    v8f acc0 = {0.f, 0.f, 0.f, 0.f, 0.f, 0.f, 0.f, 0.f};
    v8f acc1 = {0.f, 0.f, 0.f, 0.f, 0.f, 0.f, 0.f, 0.f};

    for (int k0 = 0; k0 < K; k0 += BK) {
        // ---- stage A tile (BM x BK) ----
#if USE_TDM
        if (wave == 0) {
            unsigned lds_a = (unsigned)(unsigned long long)(void*)As;
            unsigned long long ga =
                (unsigned long long)(const void*)(A + (long long)bm * K + k0);
            unsigned remK = (unsigned)(K - k0);      // remaining extents from tile
            unsigned remM = (unsigned)(M - bm);      // origin -> HW zero-fills OOB
            u32x4 g0;
            g0.x = 1u;                                             // count=1 (user D#)
            g0.y = lds_a;                                          // lds_addr
            g0.z = (unsigned)ga;                                   // global_addr lo
            g0.w = ((unsigned)(ga >> 32) & 0x01FFFFFFu) | 0x80000000u; // addr hi | type=2
            i32x8 g1 = {0, 0, 0, 0, 0, 0, 0, 0};
            g1[0] = 0x00010000;                                    // data_size = 2B
            g1[1] = (int)((remK & 0xFFFFu) << 16);                 // tensor_dim0 lo16
            g1[2] = (int)((remK >> 16) | ((remM & 0xFFFFu) << 16));// dim0 hi | dim1 lo
            g1[3] = (int)((remM >> 16) | ((unsigned)BK << 16));    // dim1 hi | tile_dim0
            g1[4] = BM;                                            // tile_dim1 (tile_dim2=0)
            g1[5] = K;                                             // tensor_dim0_stride lo32
            g1[6] = 0;
            g1[7] = 0;
            i32x4 g2 = {0, 0, 0, 0};
            i32x4 g3 = {0, 0, 0, 0};
#if __clang_major__ >= 23
            i32x8 g4 = {0, 0, 0, 0, 0, 0, 0, 0};
            __builtin_amdgcn_tensor_load_to_lds(g0, g1, g2, g3, g4, 0);
#else
            __builtin_amdgcn_tensor_load_to_lds(g0, g1, g2, g3, 0);
#endif
            __builtin_amdgcn_s_wait_tensorcnt(0);
        }
#else
        if (bm + BM <= M && k0 + BK <= K) {
            int row = tid >> 2, kf = (tid & 3) << 3;               // one uint4 / thread
            *(uint4*)&As[row * BK + kf] =
                *(const uint4*)&A[(long long)(bm + row) * K + k0 + kf];
        } else {
            for (int idx = tid; idx < BM * BK; idx += 256) {
                int row = idx >> 5, kk = idx & 31;
                int gr = bm + row, gk = k0 + kk;
                As[idx] = (gr < M && gk < K) ? A[(long long)gr * K + gk] : (bf16)0.f;
            }
        }
#endif
        // ---- stage B tile (BK x BN), transposed into [n][k] ----
        if (bn + BN <= N && k0 + BK <= K) {
            int k = tid >> 3, n8 = (tid & 7) << 3;
            union { uint4 q; bf16 h[8]; } t;
            t.q = *(const uint4*)&Bp[(long long)(k0 + k) * N + bn + n8];
            #pragma unroll
            for (int j = 0; j < 8; ++j) Bt[(n8 + j) * BK + k] = t.h[j];
        } else {
            for (int idx = tid; idx < BK * BN; idx += 256) {
                int kk = idx >> 6, n = idx & 63;
                int gk = k0 + kk, gn = bn + n;
                Bt[n * BK + kk] = (gk < K && gn < N) ? Bp[(long long)gk * N + gn] : (bf16)0.f;
            }
        }
        if (k0 + BK < K)
            __builtin_prefetch(&Bp[(long long)(k0 + BK + (tid >> 3)) * N + bn], 0, 1);
        __syncthreads();

        // ---- fragments (CDNA5 16-bit operand lane layout) + 2 WMMAs ----
        union { v16bf v; unsigned u[8]; } a0f, a1f, bf0;
        #pragma unroll
        for (int r = 0; r < 8; ++r) {
            int kk = hi16 * 8 + ((r < 4) ? (2 * r) : (16 + 2 * (r - 4)));
            a0f.u[r] = *(const unsigned*)&As[(wm * 32 +      l15) * BK + kk];
            a1f.u[r] = *(const unsigned*)&As[(wm * 32 + 16 + l15) * BK + kk];
            bf0.u[r] = *(const unsigned*)&Bt[(wn * 16 +      l15) * BK + kk];
        }
        acc0 = __builtin_amdgcn_wmma_f32_16x16x32_bf16(
            false, a0f.v, false, bf0.v, (short)0, acc0, false, false);
        acc1 = __builtin_amdgcn_wmma_f32_16x16x32_bf16(
            false, a1f.v, false, bf0.v, (short)0, acc1, false, false);
        __syncthreads();
    }

    // ---- epilogue + store (f32 C layout: hi half-wave owns rows 8..15) ----
    int colg = bn + wn * 16 + l15;
    #pragma unroll
    for (int s = 0; s < 2; ++s) {
        v8f accv = s ? acc1 : acc0;
        #pragma unroll
        for (int r = 0; r < 8; ++r) {
            int rowg = bm + wm * 32 + s * 16 + hi16 * 8 + r;
            if (rowg < M && colg < N) {
                float v = accv[r];
                if (bias) v += bias[rowg];
                if (act == 1)      v = gelu_f(v);
                else if (act == 2) v = (v > 20.f) ? v : log1pf(expf(v));  // softplus
                else if (act == 3) v = fmaxf(v, 0.f);
                long long ci = coff + (long long)rowg * N + colg;
                if (res) v += res[ci];
                if (C)  C[ci]  = v;
                if (Cb) Cb[ci] = (bf16)v;
            }
        }
    }
}

// ---------------------------------------------------------------------------
// LayerNorm over channel dim of NCHW; optional GELU; optional bf16 output.
// ---------------------------------------------------------------------------
__global__ void __launch_bounds__(256) ln2d_kernel(
    const float* __restrict__ x, const float* __restrict__ g, const float* __restrict__ b,
    float* __restrict__ out, bf16* __restrict__ outb, int Bn, int C, int HW, int act)
{
    long long i = (long long)blockIdx.x * 256 + threadIdx.x;
    long long total = (long long)Bn * HW;
    if (i >= total) return;
    long long bi = i / HW, hw = i % HW;
    const float* xp = x + (bi * C) * HW + hw;
    float mu = 0.f;
    for (int c = 0; c < C; ++c) mu += xp[(long long)c * HW];
    mu *= (1.f / C);
    float var = 0.f;
    for (int c = 0; c < C; ++c) { float d = xp[(long long)c * HW] - mu; var += d * d; }
    var *= (1.f / C);
    float inv = rsqrtf(var + LN_EPS);
    for (int c = 0; c < C; ++c) {
        float v = (xp[(long long)c * HW] - mu) * inv * g[c] + b[c];
        if (act == 1) v = gelu_f(v);
        long long o = (bi * C + c) * HW + hw;
        if (out)  out[o]  = v;
        if (outb) outb[o] = (bf16)v;
    }
}

// direct 3x3 conv, pad 1, arbitrary stride, +bias (patch embed / downsample)
__global__ void __launch_bounds__(256) conv3x3_kernel(
    const float* __restrict__ x, const float* __restrict__ w, const float* __restrict__ bias,
    float* __restrict__ out, int Bn, int Ci, int Co, int H, int W, int Ho, int Wo, int stride)
{
    long long i = (long long)blockIdx.x * 256 + threadIdx.x;
    long long total = (long long)Bn * Co * Ho * Wo;
    if (i >= total) return;
    int wo = (int)(i % Wo); long long t = i / Wo;
    int ho = (int)(t % Ho); t /= Ho;
    int o  = (int)(t % Co); int b = (int)(t / Co);
    float s = bias ? bias[o] : 0.f;
    for (int c = 0; c < Ci; ++c) {
        const float* xp = x + (((long long)b * Ci + c) * H) * W;
        const float* wp = w + ((long long)o * Ci + c) * 9;
        for (int ki = 0; ki < 3; ++ki) {
            int hh = ho * stride + ki - 1;
            if (hh < 0 || hh >= H) continue;
            for (int kj = 0; kj < 3; ++kj) {
                int ww = wo * stride + kj - 1;
                if (ww < 0 || ww >= W) continue;
                s += xp[(long long)hh * W + ww] * wp[ki * 3 + kj];
            }
        }
    }
    out[i] = s;
}

// depthwise 3x3, pad 1, stride 1, fused SiLU
__global__ void __launch_bounds__(256) dwconv_silu_kernel(
    const float* __restrict__ x, const float* __restrict__ w, float* __restrict__ out,
    int Bn, int C, int H, int W)
{
    long long i = (long long)blockIdx.x * 256 + threadIdx.x;
    long long total = (long long)Bn * C * H * W;
    if (i >= total) return;
    int wd = (int)(i % W); long long t = i / W;
    int h  = (int)(t % H); t /= H;
    int c  = (int)(t % C); int b = (int)(t / C);
    const float* xp = x + (((long long)b * C + c) * H) * W;
    const float* wp = w + (long long)c * 9;
    float s = 0.f;
    for (int ki = 0; ki < 3; ++ki) {
        int hh = h + ki - 1; if (hh < 0 || hh >= H) continue;
        for (int kj = 0; kj < 3; ++kj) {
            int ww = wd + kj - 1; if (ww < 0 || ww >= W) continue;
            s += xp[(long long)hh * W + ww] * wp[ki * 3 + kj];
        }
    }
    out[i] = s / (1.f + expf(-s));   // silu
}

// cross-scan: x[B,D,H,W] -> xs[B,4,D,L] (f32 + bf16)
__global__ void __launch_bounds__(256) cross_scan_kernel(
    const float* __restrict__ x, float* __restrict__ xs, bf16* __restrict__ xsb,
    int Bn, int D, int H, int W)
{
    long long i = (long long)blockIdx.x * 256 + threadIdx.x;
    int L = H * W;
    long long total = (long long)Bn * D * L;
    if (i >= total) return;
    int l = (int)(i % L); long long bd = i / L;
    int d = (int)(bd % D); int b = (int)(bd / D);
    int h = l / W, wd = l % W;
    int lt = wd * H + h;
    float v = x[i];
    long long DL = (long long)D * L;
    long long base = (((long long)b * 4) * D + d) * L;
    long long i0 = base + l;
    long long i1 = base + DL + lt;
    long long i2 = base + 2 * DL + (L - 1 - l);
    long long i3 = base + 3 * DL + (L - 1 - lt);
    xs[i0] = v; xsb[i0] = (bf16)v;
    xs[i1] = v; xsb[i1] = (bf16)v;
    xs[i2] = v; xsb[i2] = (bf16)v;
    xs[i3] = v; xsb[i3] = (bf16)v;
}

// cross-merge: ys[B,4,D,L] -> out[B,D,H,W]
__global__ void __launch_bounds__(256) cross_merge_kernel(
    const float* __restrict__ ys, float* __restrict__ out, int Bn, int D, int H, int W)
{
    long long i = (long long)blockIdx.x * 256 + threadIdx.x;
    int L = H * W;
    long long total = (long long)Bn * D * L;
    if (i >= total) return;
    int l = (int)(i % L); long long bd = i / L;
    int d = (int)(bd % D); int b = (int)(bd / D);
    int h = l / W, wd = l % W;
    int lt = wd * H + h;
    long long DL = (long long)D * L;
    long long base = (((long long)b * 4) * D + d) * L;
    out[i] = ys[base + l] + ys[base + 2 * DL + (L - 1 - l)]
           + ys[base + DL + lt] + ys[base + 3 * DL + (L - 1 - lt)];
}

// selective scan (first-order linear recurrence), one block per (b,k,d)
// sequence; block-wide Hillis-Steele over (A,B) affine composition in LDS.
__global__ void __launch_bounds__(256) scan_kernel(
    const float* __restrict__ dt, const float* __restrict__ A_log,
    const float* __restrict__ xdbl, const float* __restrict__ xs,
    const float* __restrict__ Ds, float* __restrict__ ys,
    int Dd, int L, int R, int Nst, int Cxp)
{
    int s = blockIdx.x;
    int d = s % Dd; int k = (s / Dd) & 3; int b = s / (Dd * 4);
    long long base = (((long long)b * 4 + k) * Dd + d) * (long long)L;
    long long xb   = ((long long)b * 4 + k) * (long long)Cxp * L;
    int tid = threadIdx.x;
    int chunk = (L + 255) / 256;     // <= 16 for L <= 4096
    float yacc[16], la[16], lb[16];
    for (int j = 0; j < 16; ++j) yacc[j] = 0.f;
    __shared__ float sA[256], sB[256];

    for (int n = 0; n < Nst; ++n) {
        float Ac = -expf(A_log[((long long)k * Dd + d) * Nst + n]);
        float aprod = 1.f, hpart = 0.f;
        for (int j = 0; j < chunk; ++j) {
            int l = tid * chunk + j;
            float a = 1.f, bx = 0.f;
            if (l < L) {
                float dv = dt[base + l];
                float Bv = xdbl[xb + (long long)(R + n) * L + l];
                float xv = xs[base + l];
                a  = expf(dv * Ac);
                bx = dv * Bv * xv;
            }
            hpart = a * hpart + bx;
            aprod *= a;
            la[j] = a; lb[j] = bx;
        }
        sA[tid] = aprod; sB[tid] = hpart;
        __syncthreads();
        for (int off = 1; off < 256; off <<= 1) {
            float pa = 1.f, pb = 0.f;
            if (tid >= off) { pa = sA[tid - off]; pb = sB[tid - off]; }
            float ca = sA[tid], cb = sB[tid];
            __syncthreads();
            sA[tid] = ca * pa;          // compose earlier-then-current
            sB[tid] = ca * pb + cb;
            __syncthreads();
        }
        float h = (tid == 0) ? 0.f : sB[tid - 1];   // exclusive carry-in
        for (int j = 0; j < chunk; ++j) {
            int l = tid * chunk + j;
            h = la[j] * h + lb[j];
            if (l < L) {
                float Cv = xdbl[xb + (long long)(R + Nst + n) * L + l];
                yacc[j] += h * Cv;
            }
        }
        __syncthreads();
    }
    float Dv = Ds[k * Dd + d];
    for (int j = 0; j < chunk; ++j) {
        int l = tid * chunk + j;
        if (l < L) ys[base + l] = yacc[j] + Dv * xs[base + l];
    }
}

__global__ void __launch_bounds__(256) bn_relu_kernel(
    const float* __restrict__ x, const float* __restrict__ g, const float* __restrict__ b,
    const float* __restrict__ m, const float* __restrict__ v, float* __restrict__ out,
    long long total, int C, int HW)
{
    long long i = (long long)blockIdx.x * 256 + threadIdx.x;
    if (i >= total) return;
    int c = (int)((i / HW) % C);
    float val = (x[i] - m[c]) * rsqrtf(v[c] + LN_EPS) * g[c] + b[c];
    out[i] = fmaxf(val, 0.f);
}

__global__ void __launch_bounds__(256) f2b_kernel(
    const float* __restrict__ x, bf16* __restrict__ y, long long n)
{
    long long i = (long long)blockIdx.x * 256 + threadIdx.x;
    if (i < n) y[i] = (bf16)x[i];
}

// gather the R leading rows of each (Cxp x L) matrix into compact bf16
__global__ void __launch_bounds__(256) f2b_rows_kernel(
    const float* __restrict__ x, bf16* __restrict__ y,
    long long total, long long msz, long long instride)
{
    long long i = (long long)blockIdx.x * 256 + threadIdx.x;
    if (i >= total) return;
    long long mm = i / msz, r = i % msz;
    y[i] = (bf16)x[mm * instride + r];
}

// ---------------------------------------------------------------------------
extern "C" void kernel_launch(void* const* d_in, const int* in_sizes, int n_in,
                              void* d_out, int out_size, void* d_ws, size_t ws_size,
                              hipStream_t stream)
{
    (void)in_sizes; (void)n_in; (void)out_size; (void)ws_size;
    int idx = 0;
    auto nx = [&]() -> const float* { return (const float*)d_in[idx++]; };

    const float* x_in = nx();
    const float *pe_w1 = nx(), *pe_b1 = nx(), *pe_g1 = nx(), *pe_be1 = nx();
    const float *pe_w2 = nx(), *pe_b2 = nx(), *pe_g2 = nx(), *pe_be2 = nx();

    const int dims[4]   = {96, 192, 384, 768};
    const int depths[4] = {2, 2, 8, 2};
    const int outc[4]   = {128, 256, 512, 512};

    struct SSM { const float *in_w,*conv_w,*xproj_w,*dtproj_w,*dtproj_b,*A_log,*Ds,*on_g,*on_b,*out_w; };
    struct Blk { const float *n1g,*n1b; SSM s; const float *n2g,*n2b,*m_w1,*m_b1,*m_w2,*m_b2; };
    struct Stg { Blk blk[8]; const float *dw,*db,*dg,*dbe,*og,*ob; };
    struct Prj { const float *w,*g,*b,*m,*v; };
    Stg st[4]; Prj pj[4];

    for (int i = 0; i < 4; ++i) {
        for (int j = 0; j < depths[i]; ++j) {
            Blk& bk = st[i].blk[j];
            bk.n1g = nx(); bk.n1b = nx();
            bk.s.in_w = nx(); bk.s.conv_w = nx(); bk.s.xproj_w = nx();
            bk.s.dtproj_w = nx(); bk.s.dtproj_b = nx(); bk.s.A_log = nx();
            bk.s.Ds = nx(); bk.s.on_g = nx(); bk.s.on_b = nx(); bk.s.out_w = nx();
            bk.n2g = nx(); bk.n2b = nx();
            bk.m_w1 = nx(); bk.m_b1 = nx(); bk.m_w2 = nx(); bk.m_b2 = nx();
        }
        if (i < 3) { st[i].dw = nx(); st[i].db = nx(); st[i].dg = nx(); st[i].dbe = nx(); }
        else       { st[i].dw = st[i].db = st[i].dg = st[i].dbe = nullptr; }
        st[i].og = nx(); st[i].ob = nx();
    }
    for (int i = 0; i < 4; ++i) { pj[i].w = nx(); pj[i].g = nx(); pj[i].b = nx(); pj[i].m = nx(); pj[i].v = nx(); }

    // ---- workspace carve (stage-1 maxima, reused across stages) ----
    char* wbase = (char*)d_ws;
    size_t woff = 0;
    auto alloc = [&](size_t bytes) -> void* {
        void* p = wbase + woff;
        woff += (bytes + 255) & ~(size_t)255;
        return p;
    };
    const size_t MAXD = 3145728ULL;    // 8*96*64*64
    const size_t MAXH = 12582912ULL;   // 8*384*4096 (hidden / xs / dt / ys)
    float* bufX = (float*)alloc(MAXD * 4);
    float* bufA = (float*)alloc(MAXD * 4);
    float* bufB = (float*)alloc(MAXD * 4);
    float* bufH = (float*)alloc(MAXH * 4);
    float* xsb  = (float*)alloc(MAXH * 4);
    float* dtb  = (float*)alloc(MAXH * 4);
    float* ysb  = (float*)alloc(MAXH * 4);
    float* xdbl = (float*)alloc(1048576ULL * 4);     // 8*4*Cxp*L max
    bf16* hA    = (bf16*)alloc(MAXD * 2);
    bf16* hH    = (bf16*)alloc(MAXH * 2);
    bf16* xs_h  = (bf16*)alloc(MAXH * 2);
    bf16* dts_h = (bf16*)alloc(786432ULL * 2);       // 8*4*R*L max
    bf16* wbf   = (bf16*)alloc(2359296ULL * 2);      // 768*3072 max weight

    const int Bn = 8;
    auto elws = [&](long long n) { return dim3((unsigned)((n + 255) / 256)); };
    auto cvt  = [&](const float* src, bf16* dst, long long n) {
        f2b_kernel<<<elws(n), 256, 0, stream>>>(src, dst, n);
    };
    auto gemm = [&](const bf16* A, const bf16* Bp, float* C, bf16* Cb,
                    const float* bias, const float* res, int M, int N, int K,
                    long long sB, long long sC, int nb, int act) {
        dim3 g((N + BN - 1) / BN, (M + BM - 1) / BM, nb);
        gemm_wmma_kernel<<<g, 256, 0, stream>>>(A, Bp, C, Cb, bias, res, M, N, K, sB, sC, act);
    };
    auto ln = [&](const float* x, const float* g, const float* b, float* o, bf16* ob,
                  int C, int HW, int act) {
        ln2d_kernel<<<elws((long long)Bn * HW), 256, 0, stream>>>(x, g, b, o, ob, Bn, C, HW, act);
    };

    // ---- patch embed ----
    conv3x3_kernel<<<elws((long long)Bn * 48 * 128 * 128), 256, 0, stream>>>(
        x_in, pe_w1, pe_b1, bufA, Bn, 3, 48, 256, 256, 128, 128, 2);
    ln(bufA, pe_g1, pe_be1, bufB, nullptr, 48, 128 * 128, 1 /*gelu*/);
    conv3x3_kernel<<<elws((long long)Bn * 96 * 64 * 64), 256, 0, stream>>>(
        bufB, pe_w2, pe_b2, bufX, Bn, 48, 96, 128, 128, 64, 64, 2);
    ln(bufX, pe_g2, pe_be2, bufX, nullptr, 96, 64 * 64, 0);

    // ---- stages ----
    int H = 64, W = 64;
    long long outOff = 0;
    for (int si = 0; si < 4; ++si) {
        int d = dims[si];
        int L = H * W;
        int R = d / 16, Nst = 1, Cxp = R + 2 * Nst;
        long long dL = (long long)d * L;

        for (int bi = 0; bi < depths[si]; ++bi) {
            const Blk& bk = st[si].blk[bi];
            // --- SS2D branch ---
            ln(bufX, bk.n1g, bk.n1b, nullptr, hA, d, L, 0);
            cvt(bk.s.in_w, wbf, (long long)d * d);
            gemm(wbf, hA, bufA, nullptr, nullptr, nullptr, d, L, d, dL, dL, Bn, 0);
            dwconv_silu_kernel<<<elws((long long)Bn * dL), 256, 0, stream>>>(bufA, bk.s.conv_w, bufB, Bn, d, H, W);
            cross_scan_kernel<<<elws((long long)Bn * dL), 256, 0, stream>>>(bufB, xsb, xs_h, Bn, d, H, W);
            cvt(bk.s.xproj_w, wbf, (long long)4 * Cxp * d);
            for (int k = 0; k < 4; ++k)
                gemm(wbf + (long long)k * Cxp * d, xs_h + (long long)k * dL,
                     xdbl + (long long)k * Cxp * L, nullptr, nullptr, nullptr,
                     Cxp, L, d, 4 * dL, (long long)4 * Cxp * L, Bn, 0);
            f2b_rows_kernel<<<elws((long long)Bn * 4 * R * L), 256, 0, stream>>>(
                xdbl, dts_h, (long long)Bn * 4 * R * L, (long long)R * L, (long long)Cxp * L);
            cvt(bk.s.dtproj_w, wbf, (long long)4 * d * R);
            for (int k = 0; k < 4; ++k)
                gemm(wbf + (long long)k * d * R, dts_h + (long long)k * R * L,
                     dtb + (long long)k * dL, nullptr, bk.s.dtproj_b + k * d, nullptr,
                     d, L, R, (long long)4 * R * L, 4 * dL, Bn, 2 /*softplus*/);
            scan_kernel<<<dim3(Bn * 4 * d), 256, 0, stream>>>(
                dtb, bk.s.A_log, xdbl, xsb, bk.s.Ds, ysb, d, L, R, Nst, Cxp);
            cross_merge_kernel<<<elws((long long)Bn * dL), 256, 0, stream>>>(ysb, bufB, Bn, d, H, W);
            ln(bufB, bk.s.on_g, bk.s.on_b, nullptr, hA, d, L, 0);
            cvt(bk.s.out_w, wbf, (long long)d * d);
            gemm(wbf, hA, bufX, nullptr, nullptr, bufX /*residual*/, d, L, d, dL, dL, Bn, 0);
            // --- MLP branch ---
            ln(bufX, bk.n2g, bk.n2b, nullptr, hA, d, L, 0);
            cvt(bk.m_w1, wbf, (long long)4 * d * d);
            gemm(wbf, hA, nullptr, hH, bk.m_b1, nullptr, 4 * d, L, d, dL, 4 * dL, Bn, 1 /*gelu*/);
            cvt(bk.m_w2, wbf, (long long)4 * d * d);
            gemm(wbf, hH, bufX, nullptr, bk.m_b2, bufX /*residual*/, d, L, 4 * d, 4 * dL, dL, Bn, 0);
        }

        // --- stage feature -> projection head -> d_out ---
        ln(bufX, st[si].og, st[si].ob, nullptr, hA, d, L, 0);
        cvt(pj[si].w, wbf, (long long)outc[si] * d);
        gemm(wbf, hA, bufH, nullptr, nullptr, nullptr, outc[si], L, d, dL,
             (long long)outc[si] * L, Bn, 0);
        bn_relu_kernel<<<elws((long long)Bn * outc[si] * L), 256, 0, stream>>>(
            bufH, pj[si].g, pj[si].b, pj[si].m, pj[si].v,
            (float*)d_out + outOff, (long long)Bn * outc[si] * L, outc[si], L);
        outOff += (long long)Bn * outc[si] * L;

        // --- downsample ---
        if (si < 3) {
            int d2 = dims[si + 1], H2 = H / 2, W2 = W / 2;
            conv3x3_kernel<<<elws((long long)Bn * d2 * H2 * W2), 256, 0, stream>>>(
                bufX, st[si].dw, st[si].db, bufB, Bn, d, d2, H, W, H2, W2, 2);
            ln(bufB, st[si].dg, st[si].dbe, bufX, nullptr, d2, H2 * W2, 0);
            H = H2; W = W2;
        }
    }
}
